// Attention_4853313044610
// MI455X (gfx1250) — matrix operand order repaired and checked by reading, not compiled, not run on hardware
//
#include <hip/hip_runtime.h>
#include <hip/hip_bf16.h>
#include <cstdint>

// ---------- types ----------
typedef __attribute__((ext_vector_type(16))) __bf16 v16bf;
typedef __attribute__((ext_vector_type(8)))  __bf16 v8bf;
typedef __attribute__((ext_vector_type(8)))  float  v8f;

#define WMMA_BF16(a, b, c) \
  __builtin_amdgcn_wmma_f32_16x16x32_bf16(false, (a), false, (b), (short)0, (c), false, false)

// ---------- CDNA5 async global->LDS copy (ASYNCcnt path, no VGPR bounce) ----------
__device__ __forceinline__ void async_copy16(const __bf16* gsrc, __bf16* ldst) {
  unsigned long long ga = (unsigned long long)(uintptr_t)gsrc;
  unsigned int       la = (unsigned int)(uintptr_t)ldst;  // low 32 bits = LDS offset
  asm volatile("global_load_async_to_lds_b128 %0, %1, off"
               :: "v"(la), "v"(ga) : "memory");
}
__device__ __forceinline__ void wait_async0() {
  asm volatile("s_wait_asynccnt 0x0" ::: "memory");
}

// ---------- DPP16 butterfly reductions over the 16-lane N group ----------
// All four controls stay within a DPP row (16 lanes), preserving half separation.
template <int CTRL>
__device__ __forceinline__ float dpp_movf(float v) {
  return __int_as_float(
      __builtin_amdgcn_update_dpp(0, __float_as_int(v), CTRL, 0xF, 0xF, true));
}
__device__ __forceinline__ float red_max16(float v) {
  v = fmaxf(v, dpp_movf<0xB1>(v));   // quad_perm [1,0,3,2]  (xor1)
  v = fmaxf(v, dpp_movf<0x4E>(v));   // quad_perm [2,3,0,1]  (xor2)
  v = fmaxf(v, dpp_movf<0x141>(v));  // row_half_mirror      (other quad in 8)
  v = fmaxf(v, dpp_movf<0x140>(v));  // row_mirror           (other 8 in 16)
  return v;
}
__device__ __forceinline__ float red_sum16(float v) {
  v += dpp_movf<0xB1>(v);
  v += dpp_movf<0x4E>(v);
  v += dpp_movf<0x141>(v);
  v += dpp_movf<0x140>(v);
  return v;
}

// ---------- fragment loaders (LDS, bf16) ----------
__device__ __forceinline__ v16bf load_a_frag(const __bf16* p, int stride) {
  const int lane = threadIdx.x & 31;
  const int row  = lane & 15;
  const int hl   = lane >> 4;
  const __bf16* q = p + row * stride + hl * 16;
  v8bf lo = *(const v8bf*)(q);
  v8bf hi = *(const v8bf*)(q + 8);
  v16bf a;
#pragma unroll
  for (int e = 0; e < 8; ++e) { a[e] = lo[e]; a[e + 8] = hi[e]; }
  return a;
}
__device__ __forceinline__ v16bf load_b_frag(const __bf16* p, int stride) {
  const int lane = threadIdx.x & 31;
  const int col  = lane & 15;
  const int hl   = lane >> 4;
  return *(const v16bf*)(p + col * stride + hl * 16);
}

// ---------- fp32 -> bf16 convert ----------
__global__ __launch_bounds__(256) void cvt_f32_bf16(const float* __restrict__ in,
                                                    __bf16* __restrict__ out, int n4) {
  int i = blockIdx.x * 256 + threadIdx.x;
  if (i < n4) {
    float4 v = *(const float4*)(in + (size_t)i * 4);
    __bf16* o = out + (size_t)i * 4;
    o[0] = (__bf16)v.x; o[1] = (__bf16)v.y; o[2] = (__bf16)v.z; o[3] = (__bf16)v.w;
  }
}

// ---------- GEMM: C[M,N] = A[M,K] * W[N,K]^T ----------
// 128x128 block tile, BK=64, double-buffered async-LDS pipeline,
// 8 waves (4Mx2N), wave tile 32x64 => 16 WMMA per stage, 1 barrier per stage.
template <int OUT_F32>
__global__ __launch_bounds__(256) void gemm_bf16_wmma(
    const __bf16* __restrict__ A, const __bf16* __restrict__ W,
    void* __restrict__ Cout, const float* __restrict__ bias,
    int M, int N, int K) {
  __shared__ __align__(32) __bf16 sA[2][128 * 80];
  __shared__ __align__(32) __bf16 sB[2][128 * 80];

  const int tid  = threadIdx.x;
  const int lane = tid & 31;
  const int wave = tid >> 5;
  const int wm   = (wave & 3) * 32;
  const int wn   = (wave >> 2) * 64;
  const int m0   = blockIdx.y * 128;
  const int n0   = blockIdx.x * 128;
  const int col16 = lane & 15, hl = lane >> 4;

  v8f acc[2][4];
#pragma unroll
  for (int mi = 0; mi < 2; ++mi)
#pragma unroll
    for (int ni = 0; ni < 4; ++ni)
#pragma unroll
      for (int e = 0; e < 8; ++e) acc[mi][ni][e] = 0.f;

  auto stage = [&](int k0, int buf) {
#pragma unroll
    for (int i = 0; i < 4; ++i) {
      int c = tid + i * 256;
      int r = c >> 3, kc = (c & 7) * 8;
      async_copy16(&A[(size_t)(m0 + r) * K + k0 + kc], &sA[buf][r * 80 + kc]);
      async_copy16(&W[(size_t)(n0 + r) * K + k0 + kc], &sB[buf][r * 80 + kc]);
    }
  };

  const int NKT = K / 64;
  stage(0, 0);
  for (int kt = 0; kt < NKT; ++kt) {
    const int buf = kt & 1;
    wait_async0();
    __syncthreads();
    if (kt + 1 < NKT) stage((kt + 1) * 64, buf ^ 1);
#pragma unroll
    for (int ks = 0; ks < 2; ++ks) {
      v16bf af[2], bfm[4];
      af[0] = load_a_frag(&sA[buf][(wm + 0)  * 80 + ks * 32], 80);
      af[1] = load_a_frag(&sA[buf][(wm + 16) * 80 + ks * 32], 80);
#pragma unroll
      for (int ni = 0; ni < 4; ++ni)
        bfm[ni] = load_b_frag(&sB[buf][(wn + ni * 16) * 80 + ks * 32], 80);
#pragma unroll
      for (int mi = 0; mi < 2; ++mi)
#pragma unroll
        for (int ni = 0; ni < 4; ++ni)
          acc[mi][ni] = WMMA_BF16(af[mi], bfm[ni], acc[mi][ni]);
    }
  }

#pragma unroll
  for (int mi = 0; mi < 2; ++mi) {
#pragma unroll
    for (int ni = 0; ni < 4; ++ni) {
      int col   = n0 + wn + ni * 16 + col16;
      int rbase = m0 + wm + mi * 16 + hl * 8;
      if (OUT_F32) {
        float bv = bias[col];
        float* C = (float*)Cout;
#pragma unroll
        for (int i = 0; i < 8; ++i)
          C[(size_t)(rbase + i) * N + col] = acc[mi][ni][i] + bv;
      } else {
        __bf16* C = (__bf16*)Cout;
#pragma unroll
        for (int i = 0; i < 8; ++i)
          C[(size_t)(rbase + i) * N + col] = (__bf16)acc[mi][ni][i];
      }
    }
  }
}

// ---------- flash attention ----------
// Q,K,V,O bf16 as [B*S, H*D] = [4096, 1024]. Grid: (B*H, S/128).
#define SEQ 2048
#define EMB 1024

__global__ __launch_bounds__(256) void flash_attn_wmma(
    const __bf16* __restrict__ Q, const __bf16* __restrict__ Kg,
    const __bf16* __restrict__ Vg, __bf16* __restrict__ O) {
  __shared__ __align__(32) __bf16 sK [64 * 80];       // [key][d]
  __shared__ __align__(32) __bf16 sVt[64 * 80];       // [d][key]
  __shared__ __align__(32) __bf16 sP [8 * 16 * 64];   // per-wave P tile

  const int tid = threadIdx.x, lane = tid & 31, wave = tid >> 5;
  const int col16 = lane & 15, hl = lane >> 4;
  const int b = blockIdx.x >> 4, h = blockIdx.x & 15;
  const int q0 = blockIdx.y * 128 + wave * 16;
  const size_t base = ((size_t)b * SEQ) * EMB + (size_t)h * 64;

  // stage this wave's 16x64 Q tile, pre-scaled by 1/sqrt(d)=0.125 (exact in bf16)
  __bf16* myP = &sP[wave * 16 * 64];
#pragma unroll
  for (int i = 0; i < 4; ++i) {
    int c = lane + i * 32;
    int r = c >> 3, dc = (c & 7) * 8;
    v8bf qv = *(const v8bf*)&Q[base + (size_t)(q0 + r) * EMB + dc];
#pragma unroll
    for (int e = 0; e < 8; ++e) qv[e] = (__bf16)((float)qv[e] * 0.125f);
    *(v8bf*)&myP[r * 64 + dc] = qv;
  }
  asm volatile("s_wait_dscnt 0x0" ::: "memory");
  v16bf qf[2];
  qf[0] = load_a_frag(myP, 64);
  qf[1] = load_a_frag(myP + 32, 64);

  float mrow[8], lrow[8];
  v8f acc[4];
#pragma unroll
  for (int i = 0; i < 8; ++i) { mrow[i] = -3.0e38f; lrow[i] = 0.f; }
#pragma unroll
  for (int t = 0; t < 4; ++t)
#pragma unroll
    for (int e = 0; e < 8; ++e) acc[t][e] = 0.f;

  for (int kb = 0; kb < SEQ; kb += 64) {
    __syncthreads();
    // K tile: async direct-to-LDS; V tile: register path with transpose store
#pragma unroll
    for (int i = 0; i < 2; ++i) {
      int c = tid + i * 256;
      int r = c >> 3, dc = (c & 7) * 8;
      async_copy16(&Kg[base + (size_t)(kb + r) * EMB + dc], &sK[r * 80 + dc]);
      v8bf vv = *(const v8bf*)&Vg[base + (size_t)(kb + r) * EMB + dc];
#pragma unroll
      for (int e = 0; e < 8; ++e) sVt[(dc + e) * 80 + r] = vv[e];
    }
    wait_async0();
    __syncthreads();

    // S = (Q/sqrt(d)) K^T  (scale already folded into Q)
    v8f st[4];
#pragma unroll
    for (int t = 0; t < 4; ++t) {
#pragma unroll
      for (int e = 0; e < 8; ++e) st[t][e] = 0.f;
#pragma unroll
      for (int ks = 0; ks < 2; ++ks) {
        v16bf bk = load_b_frag(&sK[(t * 16) * 80 + ks * 32], 80);
        st[t] = WMMA_BF16(qf[ks], bk, st[t]);
      }
    }

    // online softmax: DPP16 butterflies over the 16-lane N group (no LDS, no waits)
    float corr[8], rs[8];
#pragma unroll
    for (int i = 0; i < 8; ++i) {
      float tm = fmaxf(fmaxf(st[0][i], st[1][i]), fmaxf(st[2][i], st[3][i]));
      tm = red_max16(tm);
      float mnew = fmaxf(mrow[i], tm);
      corr[i] = __expf(mrow[i] - mnew);
      mrow[i] = mnew;
      rs[i] = 0.f;
    }
#pragma unroll
    for (int t = 0; t < 4; ++t)
#pragma unroll
      for (int i = 0; i < 8; ++i) {
        float p = __expf(st[t][i] - mrow[i]);
        st[t][i] = p;
        rs[i] += p;
      }
#pragma unroll
    for (int i = 0; i < 8; ++i)
      lrow[i] = lrow[i] * corr[i] + red_sum16(rs[i]);
#pragma unroll
    for (int t = 0; t < 4; ++t)
#pragma unroll
      for (int i = 0; i < 8; ++i) acc[t][i] *= corr[i];

    // relayout P (C-format) -> LDS -> A-fragments, then O += P * V
#pragma unroll
    for (int t = 0; t < 4; ++t)
#pragma unroll
      for (int i = 0; i < 8; ++i)
        myP[(i + hl * 8) * 64 + t * 16 + col16] = (__bf16)st[t][i];
    asm volatile("s_wait_dscnt 0x0" ::: "memory");

    v16bf pf[2];
    pf[0] = load_a_frag(myP, 64);
    pf[1] = load_a_frag(myP + 32, 64);
#pragma unroll
    for (int t = 0; t < 4; ++t)
#pragma unroll
      for (int ks = 0; ks < 2; ++ks) {
        v16bf bv = load_b_frag(&sVt[(t * 16) * 80 + ks * 32], 80);
        acc[t] = WMMA_BF16(pf[ks], bv, acc[t]);
      }
  }

  float inv[8];
#pragma unroll
  for (int i = 0; i < 8; ++i) inv[i] = 1.f / lrow[i];
#pragma unroll
  for (int t = 0; t < 4; ++t)
#pragma unroll
    for (int i = 0; i < 8; ++i)
      O[base + (size_t)(q0 + i + hl * 8) * EMB + t * 16 + col16] =
          (__bf16)(acc[t][i] * inv[i]);
}

// ---------- host side ----------
extern "C" void kernel_launch(void* const* d_in, const int* in_sizes, int n_in,
                              void* d_out, int out_size, void* d_ws, size_t ws_size,
                              hipStream_t stream) {
  (void)in_sizes; (void)n_in; (void)out_size; (void)ws_size;
  const float* x  = (const float*)d_in[0];
  const float* Wq = (const float*)d_in[1];
  const float* Wk = (const float*)d_in[2];
  const float* Wv = (const float*)d_in[3];
  const float* Wo = (const float*)d_in[4];
  const float* bo = (const float*)d_in[5];

  const int M = 2 * SEQ;  // 4096
  const int E = EMB;      // 1024

  char* ws = (char*)d_ws;
  __bf16* xb  = (__bf16*)(ws + 0);
  __bf16* Wqb = (__bf16*)(ws + (8u  << 20));
  __bf16* Wkb = (__bf16*)(ws + (10u << 20));
  __bf16* Wvb = (__bf16*)(ws + (12u << 20));
  __bf16* Wob = (__bf16*)(ws + (14u << 20));
  __bf16* Qb  = (__bf16*)(ws + (16u << 20));
  __bf16* Kb  = (__bf16*)(ws + (24u << 20));
  __bf16* Vb  = (__bf16*)(ws + (32u << 20));
  __bf16* Ab  = (__bf16*)(ws + (40u << 20));

  cvt_f32_bf16<<<(M * E / 4 + 255) / 256, 256, 0, stream>>>(x,  xb,  M * E / 4);
  cvt_f32_bf16<<<(E * E / 4 + 255) / 256, 256, 0, stream>>>(Wq, Wqb, E * E / 4);
  cvt_f32_bf16<<<(E * E / 4 + 255) / 256, 256, 0, stream>>>(Wk, Wkb, E * E / 4);
  cvt_f32_bf16<<<(E * E / 4 + 255) / 256, 256, 0, stream>>>(Wv, Wvb, E * E / 4);
  cvt_f32_bf16<<<(E * E / 4 + 255) / 256, 256, 0, stream>>>(Wo, Wob, E * E / 4);

  dim3 ggrid(E / 128, M / 128);
  gemm_bf16_wmma<0><<<ggrid, 256, 0, stream>>>(xb, Wqb, Qb, bo, M, E, E);
  gemm_bf16_wmma<0><<<ggrid, 256, 0, stream>>>(xb, Wkb, Kb, bo, M, E, E);
  gemm_bf16_wmma<0><<<ggrid, 256, 0, stream>>>(xb, Wvb, Vb, bo, M, E, E);

  flash_attn_wmma<<<dim3(32, SEQ / 128), 256, 0, stream>>>(Qb, Kb, Vb, Ab);

  gemm_bf16_wmma<1><<<ggrid, 256, 0, stream>>>(Ab, Wob, d_out, bo, M, E, E);
}
